// GATLayer_31009663877406
// MI455X (gfx1250) — matrix operand order repaired and checked by reading, not compiled, not run on hardware
//
#include <hip/hip_runtime.h>

// GAT layer for MI455X (gfx1250, wave32).
// Pipeline (all on `stream`, serialized by stream order):
//   K0 pack_w   : W[128x128] f32 -> f16, pre-swizzled into WMMA B-fragment layout (32 KB)
//   K1 init     : out[n,c] = bias[c]; emax keys = 0; denom = 0
//   K2 gemm     : h = x @ W via v_wmma_f32_16x16x32_f16.
//                 x-tile staged ONCE per block into LDS as f16 (8 waves share it),
//                 A fragments via ds_load_b128; B fragments from pre-packed
//                 global buffer (L1/L2-resident, 32 KB). Branch-free epilogue
//                 for full tiles.
//   K3 att      : a_src/a_dst per-node logits
//   K4 edge/max : e = leakyrelu(a_src[s]+a_dst[d]); segment-max via u32-key atomicMax
//   K5 exp/sum  : ex = exp(e - max); segment-sum via atomicAdd
//   K6 scatter  : out[d] += h[s] * alpha  (1 wave per edge, float4 per lane, f32 atomics)

#define HEADS     4
#define OUT_C     32
#define FDIM      128          // IN == HEADS*OUT_C == 128
#define NEG_SLOPE 0.2f
#define EPS_F     1e-16f
#define A_STRIDE  136          // halves per LDS row: 272 B = 16B-aligned, bank-skewed

typedef __attribute__((ext_vector_type(16))) _Float16 v16h;
typedef __attribute__((ext_vector_type(8)))  _Float16 v8h;
typedef __attribute__((ext_vector_type(8)))  float    v8f;
typedef __attribute__((ext_vector_type(4)))  float    v4f;

// ---- monotonic float<->uint key for atomicMax-based segment max ------------
__device__ __forceinline__ unsigned int f2key(float f) {
  unsigned int u = __builtin_bit_cast(unsigned int, f);
  return (u & 0x80000000u) ? ~u : (u | 0x80000000u);
}
__device__ __forceinline__ float key2f(unsigned int k) {
  unsigned int u = (k & 0x80000000u) ? (k ^ 0x80000000u) : ~k;
  return __builtin_bit_cast(float, u);
}

// ---- K0: pack W into WMMA B-fragment order --------------------------------
// B frag (16x16x32 f16): lane l -> column N = l&15; element e (0..15) -> K =
// (l>=16 ? 8 : 0) + (e&7) + (e>=8 ? 16 : 0).  Bpk[((ct*4+ks)*32+lane)*16+e] so the GEMM inner loop
// loads one contiguous 32B v16h per lane (2x global_load_b128).
__global__ void gat_pack_w(const float* __restrict__ W, _Float16* __restrict__ Bpk) {
  int t = blockIdx.x * blockDim.x + threadIdx.x;       // 8*4*32*16 = 16384
  if (t >= 8 * 4 * 32 * 16) return;
  int e    = t & 15;
  int lane = (t >> 4) & 31;
  int ks   = (t >> 9) & 3;
  int ct   = t >> 11;
  int kk   = ks * 32 + ((lane >> 4) << 3) + (e & 7) + ((e >> 3) << 4); // K index 0..127
  int col  = ct * 16 + (lane & 15);                    // N index 0..127
  Bpk[t] = (_Float16)W[kk * FDIM + col];
}

// ---- K1: init out with bias, zero softmax state ---------------------------
__global__ void gat_init(float* __restrict__ out, const float* __restrict__ bias,
                         unsigned int* __restrict__ emax_u, float* __restrict__ denom,
                         int N) {
  long long t = (long long)blockIdx.x * blockDim.x + threadIdx.x;
  long long tot = (long long)N * FDIM;
  if (t < tot) out[t] = bias[t & (FDIM - 1)];
  if (t < (long long)N * HEADS) { emax_u[t] = 0u; denom[t] = 0.0f; }
}

// ---- K2: h = x @ W via WMMA ------------------------------------------------
// Block = 256 threads = 8 waves; block owns 16 rows, wave w owns cols [16w,16w+16).
__global__ void __launch_bounds__(256) gat_gemm_wmma(
    const float* __restrict__ x, const _Float16* __restrict__ Bpk,
    float* __restrict__ h, int N) {
  __shared__ __attribute__((aligned(16))) _Float16 As[16 * A_STRIDE];

  // --- stage 16x128 x-tile into LDS once (f32 -> f16), fully coalesced -----
  {
    const int r = threadIdx.x >> 4;              // 0..15 row in tile
    const int c = (threadIdx.x & 15) * 8;        // 0..120 step 8
    const int row = blockIdx.x * 16 + r;
    const float* xp = x + (long long)(row < N ? row : 0) * FDIM + c;
    v4f p0 = *(const v4f*)(xp);                  // global_load_b128
    v4f p1 = *(const v4f*)(xp + 4);              // global_load_b128
    v8h hv;
    hv[0] = (_Float16)p0.x; hv[1] = (_Float16)p0.y;
    hv[2] = (_Float16)p0.z; hv[3] = (_Float16)p0.w;
    hv[4] = (_Float16)p1.x; hv[5] = (_Float16)p1.y;
    hv[6] = (_Float16)p1.z; hv[7] = (_Float16)p1.w;
    *(v8h*)(&As[r * A_STRIDE + c]) = hv;         // ds_store_b128
  }
  __syncthreads();

  const int wave = threadIdx.x >> 5;       // 0..7 : column tile
  const int lane = threadIdx.x & 31;
  const int hi   = lane >> 4;              // lane half-group
  const int m    = lane & 15;

  v8f acc = {};
#pragma unroll
  for (int ks = 0; ks < 4; ++ks) {
    // A frag 16x32 f16: lane (m,hi) needs K' = {0..7} and {16..23} at
    // K-base = ks*32 + hi*8  ->  two contiguous 16B LDS reads.
    const int base = m * A_STRIDE + ks * 32 + hi * 8;
    v8h lo  = *(const v8h*)(&As[base]);          // ds_load_b128 (K'=0..7)
    v8h hi8 = *(const v8h*)(&As[base + 16]);     // ds_load_b128 (K'=16..23)
    v16h a = __builtin_shufflevector(lo, hi8, 0, 1, 2, 3, 4, 5, 6, 7,
                                     8, 9, 10, 11, 12, 13, 14, 15);
    v16h b = *(const v16h*)(Bpk + ((wave * 4 + ks) * 32 + lane) * 16);
    acc = __builtin_amdgcn_wmma_f32_16x16x32_f16(
        /*neg_a=*/false, a, /*neg_b=*/false, b,
        /*c_mod=*/(short)0, acc, /*reuse_a=*/false, /*reuse_b=*/false);
  }

  // C/D layout: VGPR v -> M = v + 8*hi ; N = lane&15.
  const int r0  = blockIdx.x * 16 + hi * 8;      // first row this lane stores
  const int col = wave * 16 + m;
  float* hp = h + (long long)r0 * FDIM + col;    // rows step by FDIM floats
  if (blockIdx.x * 16 + 16 <= N) {
    // full tile: branch-free, constant-offset stores
#pragma unroll
    for (int v = 0; v < 8; ++v) hp[(long long)v * FDIM] = acc[v];
  } else {
#pragma unroll
    for (int v = 0; v < 8; ++v)
      if (r0 + v < N) hp[(long long)v * FDIM] = acc[v];
  }
}

// ---- K3: per-node attention logits ----------------------------------------
__global__ void gat_att_logits(const float* __restrict__ hbuf,
                               const float* __restrict__ att_src,
                               const float* __restrict__ att_dst,
                               float* __restrict__ a_src, float* __restrict__ a_dst,
                               int N) {
  int t = blockIdx.x * blockDim.x + threadIdx.x;
  if (t >= N * HEADS) return;
  const int n = t >> 2, hd = t & 3;
  const float* hp = hbuf + (long long)n * FDIM + hd * OUT_C;
  const float* as = att_src + hd * OUT_C;
  const float* ad = att_dst + hd * OUT_C;
  float s = 0.0f, d = 0.0f;
#pragma unroll
  for (int c = 0; c < OUT_C; c += 4) {
    v4f hv = *(const v4f*)(hp + c);
    v4f av = *(const v4f*)(as + c);
    v4f dv = *(const v4f*)(ad + c);
    s += hv.x * av.x + hv.y * av.y + hv.z * av.z + hv.w * av.w;
    d += hv.x * dv.x + hv.y * dv.y + hv.z * dv.z + hv.w * dv.w;
  }
  a_src[t] = s;
  a_dst[t] = d;
}

// ---- K4: per-edge logits + segment max ------------------------------------
__global__ void gat_edge_max(const long long* __restrict__ ei, long long E, int N,
                             const float* __restrict__ a_src,
                             const float* __restrict__ a_dst,
                             float* __restrict__ e_ws,
                             unsigned int* __restrict__ emax_u) {
  long long i = (long long)blockIdx.x * blockDim.x + threadIdx.x;
  if (i >= E + N) return;
  long long s = (i < E) ? ei[i]     : (i - E);   // self-loops appended
  long long d = (i < E) ? ei[E + i] : (i - E);
  v4f e = *(const v4f*)(a_src + s * HEADS) + *(const v4f*)(a_dst + d * HEADS);
  e.x = e.x > 0.0f ? e.x : e.x * NEG_SLOPE;
  e.y = e.y > 0.0f ? e.y : e.y * NEG_SLOPE;
  e.z = e.z > 0.0f ? e.z : e.z * NEG_SLOPE;
  e.w = e.w > 0.0f ? e.w : e.w * NEG_SLOPE;
  *(v4f*)(e_ws + i * HEADS) = e;
  unsigned int* em = emax_u + d * HEADS;
  atomicMax(em + 0, f2key(e.x));
  atomicMax(em + 1, f2key(e.y));
  atomicMax(em + 2, f2key(e.z));
  atomicMax(em + 3, f2key(e.w));
}

// ---- K5: exp(e - max) + segment sum ---------------------------------------
__global__ void gat_exp_sum(const long long* __restrict__ ei, long long E, int N,
                            float* __restrict__ e_ws,
                            const unsigned int* __restrict__ emax_u,
                            float* __restrict__ denom) {
  long long i = (long long)blockIdx.x * blockDim.x + threadIdx.x;
  if (i >= E + N) return;
  long long d = (i < E) ? ei[E + i] : (i - E);
  v4f e = *(const v4f*)(e_ws + i * HEADS);
  const unsigned int* em = emax_u + d * HEADS;
  v4f ex;
  ex.x = __expf(e.x - key2f(em[0]));
  ex.y = __expf(e.y - key2f(em[1]));
  ex.z = __expf(e.z - key2f(em[2]));
  ex.w = __expf(e.w - key2f(em[3]));
  *(v4f*)(e_ws + i * HEADS) = ex;                 // overwrite with numerator
  float* dn = denom + d * HEADS;
  atomicAdd(dn + 0, ex.x);
  atomicAdd(dn + 1, ex.y);
  atomicAdd(dn + 2, ex.z);
  atomicAdd(dn + 3, ex.w);
}

// ---- K6: weighted message scatter -----------------------------------------
// One wave32 per edge: lane handles 4 channels (32 lanes * 4 = 128 = FDIM).
__global__ void __launch_bounds__(256) gat_scatter(
    const long long* __restrict__ ei, long long E, int N,
    const float* __restrict__ hbuf, const float* __restrict__ e_ws,
    const float* __restrict__ denom, float* __restrict__ out) {
  long long w = ((long long)blockIdx.x * blockDim.x + threadIdx.x) >> 5;
  const int lane = threadIdx.x & 31;
  if (w >= E + N) return;
  long long s = (w < E) ? ei[w]     : (w - E);
  long long d = (w < E) ? ei[E + w] : (w - E);
  const int c0 = lane * 4;
  const int hd = c0 >> 5;                         // head of this lane's 4 channels
  float alpha = e_ws[w * HEADS + hd] / (denom[d * HEADS + hd] + EPS_F);
  v4f hv = *(const v4f*)(hbuf + s * FDIM + c0);   // coalesced b128 gather
  float* op = out + d * FDIM + c0;
  __builtin_prefetch(op, 1, 1);                   // global_prefetch_b8 on atomic target
  atomicAdd(op + 0, hv.x * alpha);
  atomicAdd(op + 1, hv.y * alpha);
  atomicAdd(op + 2, hv.z * alpha);
  atomicAdd(op + 3, hv.w * alpha);
}

// ---------------------------------------------------------------------------
extern "C" void kernel_launch(void* const* d_in, const int* in_sizes, int n_in,
                              void* d_out, int out_size, void* d_ws, size_t ws_size,
                              hipStream_t stream) {
  const float*     x       = (const float*)d_in[0];
  const long long* ei      = (const long long*)d_in[1];   // int64 [2,E]
  const float*     W       = (const float*)d_in[2];
  const float*     att_src = (const float*)d_in[3];
  const float*     att_dst = (const float*)d_in[4];
  const float*     bias    = (const float*)d_in[5];
  float*           out     = (float*)d_out;

  const int       N = in_sizes[0] / FDIM;
  const long long E = (long long)in_sizes[1] / 2;
  const long long T = E + N;                      // edges incl. self-loops

  // Workspace carve-out (256B aligned): h(51.2MB) Bpk(32KB) a_src/a_dst
  // emax denom (1.6MB each) e_ws(27.2MB)  ~= 83 MB total.
  char* ws = (char*)d_ws;
  size_t off = 0;
  auto carve = [&](size_t bytes) -> void* {
    void* p = ws + off;
    off = (off + bytes + 255) & ~(size_t)255;
    return p;
  };
  float*        hbuf   = (float*)carve((size_t)N * FDIM * sizeof(float));
  _Float16*     Bpk    = (_Float16*)carve(8 * 4 * 32 * 16 * sizeof(_Float16));
  float*        a_src  = (float*)carve((size_t)N * HEADS * sizeof(float));
  float*        a_dst  = (float*)carve((size_t)N * HEADS * sizeof(float));
  unsigned int* emax_u = (unsigned int*)carve((size_t)N * HEADS * sizeof(unsigned int));
  float*        denom  = (float*)carve((size_t)N * HEADS * sizeof(float));
  float*        e_ws   = (float*)carve((size_t)T * HEADS * sizeof(float));
  (void)ws_size; (void)n_in; (void)out_size;

  const int TB = 256;
  gat_pack_w<<<(8 * 4 * 32 * 16 + TB - 1) / TB, TB, 0, stream>>>(W, Bpk);

  long long initN = (long long)N * FDIM;
  gat_init<<<(unsigned)((initN + TB - 1) / TB), TB, 0, stream>>>(out, bias, emax_u, denom, N);

  gat_gemm_wmma<<<(N + 15) / 16, TB, 0, stream>>>(x, Bpk, hbuf, N);

  gat_att_logits<<<(N * HEADS + TB - 1) / TB, TB, 0, stream>>>(
      hbuf, att_src, att_dst, a_src, a_dst, N);

  gat_edge_max<<<(unsigned)((T + TB - 1) / TB), TB, 0, stream>>>(
      ei, E, N, a_src, a_dst, e_ws, emax_u);

  gat_exp_sum<<<(unsigned)((T + TB - 1) / TB), TB, 0, stream>>>(
      ei, E, N, e_ws, emax_u, denom);

  gat_scatter<<<(unsigned)((T + 7) / 8), TB, 0, stream>>>(
      ei, E, N, hbuf, e_ws, denom, out);
}
